// Stage_11244224381392
// MI455X (gfx1250) — matrix operand order repaired and checked by reading, not compiled, not run on hardware
//
#include <hip/hip_runtime.h>
#include <math.h>

// ---------------------------------------------------------------------------
// Point-cloud Mamba mixer forward for MI455X (gfx1250, wave32, WMMA, f16
// operand mirrors, async global->LDS staging).
// ---------------------------------------------------------------------------

typedef __attribute__((ext_vector_type(16))) _Float16 v16h;
typedef __attribute__((ext_vector_type(8)))  _Float16 v8h;
typedef __attribute__((ext_vector_type(8)))  float    v8f;
typedef __attribute__((ext_vector_type(4)))  int      i4v;

#define NPTS   16384
#define DMODEL 256
#define DINNER 512
#define KNBR   27
#define DEPTH  2
#define NORD   3

// ---- async global->LDS staging (gfx1250), with safe fallback --------------
#if defined(__has_builtin)
#if __has_builtin(__builtin_amdgcn_global_load_async_to_lds_b128) && \
    __has_builtin(__builtin_amdgcn_s_wait_asynccnt)
#define USE_ASYNC_LDS 1
#endif
#endif

// Copy 16 consecutive halves (32B) from global to LDS (16B aligned).
__device__ __forceinline__ void stage16h(const _Float16* __restrict__ src,
                                         _Float16* __restrict__ dst) {
#ifdef USE_ASYNC_LDS
#pragma unroll
  for (int i = 0; i < 2; ++i)
    __builtin_amdgcn_global_load_async_to_lds_b128(
        (__attribute__((address_space(1))) i4v*)
            (__attribute__((address_space(1))) void*)(src + 8 * i),
        (__attribute__((address_space(3))) i4v*)
            (__attribute__((address_space(3))) void*)(dst + 8 * i),
        0, 0);
#else
#pragma unroll
  for (int i = 0; i < 2; ++i)
    ((float4*)dst)[i] = ((const float4*)src)[i];
#endif
}
__device__ __forceinline__ void stage_wait() {
#ifdef USE_ASYNC_LDS
  __builtin_amdgcn_s_wait_asynccnt(0);
#endif
  __syncthreads();
}

__device__ __forceinline__ v16h cat8(v8h lo, v8h hi) {
  return __builtin_shufflevector(lo, hi, 0, 1, 2, 3, 4, 5, 6, 7,
                                 8, 9, 10, 11, 12, 13, 14, 15);
}

__device__ __forceinline__ float wave_sum32(float v) {
#pragma unroll
  for (int off = 16; off > 0; off >>= 1) v += __shfl_xor(v, off, 32);
  return v;
}
__device__ __forceinline__ float sigmoid_(float x) { return 1.0f / (1.0f + __expf(-x)); }

// ---------------------------------------------------------------------------
// Tiled GEMM (f16 operands, f32 accumulate): D[n,j] = sum_c A[n,c]*W[j,c]+b[j]
//   A: Nrows x KDIM row-major f16, W: Ncols x KDIM row-major f16.
// Block = 256 threads (8 waves) -> 128x64 output tile; wave -> 16 rows x 4
// column tiles (A fragment reused 4x). W chunk (64 cols x 64 K f16) staged in
// LDS via async b128 copies and shared by all 8 waves.
// ---------------------------------------------------------------------------
template <int KDIM>
__global__ void __launch_bounds__(256) gemm_tiled_h(
    const _Float16* __restrict__ A, const _Float16* __restrict__ W,
    const float* __restrict__ bias, float* __restrict__ D, int Ncols) {
  __shared__ _Float16 Bs[64][72];         // [local col][k within chunk], padded

  const int wave = threadIdx.x >> 5;
  const int lane = threadIdx.x & 31;
  const int grp  = lane >> 4;
  const int lr   = lane & 15;
  const int rowBase = blockIdx.y * 128 + wave * 16;
  const int colBase = blockIdx.x * 64;

  const _Float16* __restrict__ Arow = A + (size_t)(rowBase + lr) * KDIM;

  // staging map: thread -> (local col tj, k offset tk) ; 16 halves per thread
  const int tj = threadIdx.x >> 2;        // 0..63
  const int tk = (threadIdx.x & 3) << 4;  // 0,16,32,48
  const _Float16* __restrict__ Wst = W + (size_t)(colBase + tj) * KDIM + tk;

  v8f acc[4] = {};
  for (int kc = 0; kc < KDIM; kc += 64) {
    __syncthreads();                       // protect LDS reuse
    stage16h(Wst + kc, &Bs[tj][tk]);       // async global->LDS
    __builtin_prefetch(Arow + kc + 64, 0, 1);
    stage_wait();
#pragma unroll
    for (int ks = 0; ks < 64; ks += 32) {
      // A 16x32 f16 layout: lanes 0-15 hold K {0..7,16..23}, 16-31 {8..15,24..31}
      const v8h alo = *(const v8h*)(Arow + kc + ks + (grp << 3));
      const v8h ahi = *(const v8h*)(Arow + kc + ks + 16 + (grp << 3));
      const v16h af = cat8(alo, ahi);
#pragma unroll
      for (int ct = 0; ct < 4; ++ct) {
        const v8h blo = *(const v8h*)&Bs[ct * 16 + lr][ks + (grp << 3)];
        const v8h bhi = *(const v8h*)&Bs[ct * 16 + lr][ks + 16 + (grp << 3)];
        const v16h bf = cat8(blo, bhi);
        acc[ct] = __builtin_amdgcn_wmma_f32_16x16x32_f16(false, af, false, bf,
                                                         (short)0, acc[ct],
                                                         false, false);
      }
    }
  }
#pragma unroll
  for (int ct = 0; ct < 4; ++ct) {
    const int col = colBase + ct * 16 + lr;
    const float bv = bias ? bias[col] : 0.0f;
    float* __restrict__ Dt = D + (size_t)(rowBase + (grp << 3)) * Ncols + col;
#pragma unroll
    for (int r2 = 0; r2 < 8; ++r2) Dt[(size_t)r2 * Ncols] = acc[ct][r2] + bv;
  }
}

// ---------------------------------------------------------------------------
// CPE fused gather-GEMM on f16: t1[n,j] = cpe_b[j] + sum_k mask[n,k] *
//   sum_c feat[idx[n,k],c] * cpe_wt[k,j,c]   (cpe_w pre-transposed to [k,j,c])
// Output stored as f16 (feeds fc1 GEMM only).
// ---------------------------------------------------------------------------
__global__ void __launch_bounds__(256) cpe_gemm_h(
    const _Float16* __restrict__ feat_h, const int* __restrict__ nbr_idx,
    const int* __restrict__ nbr_mask, const _Float16* __restrict__ cpe_wt,
    const float* __restrict__ cpe_b, _Float16* __restrict__ D) {
  __shared__ _Float16 Bs[64][72];

  const int wave = threadIdx.x >> 5;
  const int lane = threadIdx.x & 31;
  const int grp = lane >> 4, lr = lane & 15;
  const int rowBase = blockIdx.y * 128 + wave * 16;
  const int colBase = blockIdx.x * 64;
  const int p = rowBase + lr;             // point (A-fragment row)

  const int tj = threadIdx.x >> 2;
  const int tk = (threadIdx.x & 3) << 4;

  const v16h zf = {};
  v8f acc[4] = {};
  for (int k = 0; k < KNBR; ++k) {
    const int idx = nbr_idx[p * KNBR + k];
    const int msk = nbr_mask[p * KNBR + k];
    const _Float16* __restrict__ g  = feat_h + (size_t)idx * DMODEL;
    const _Float16* __restrict__ Wk = cpe_wt + (size_t)k * DMODEL * DMODEL;
    const _Float16* __restrict__ Wst = Wk + (size_t)(colBase + tj) * DMODEL + tk;
    for (int kc = 0; kc < DMODEL; kc += 64) {
      __syncthreads();
      stage16h(Wst + kc, &Bs[tj][tk]);
      stage_wait();
#pragma unroll
      for (int ks = 0; ks < 64; ks += 32) {
        const v8h alo = *(const v8h*)(g + kc + ks + (grp << 3));
        const v8h ahi = *(const v8h*)(g + kc + ks + 16 + (grp << 3));
        const v16h af = msk ? cat8(alo, ahi) : zf;
#pragma unroll
        for (int ct = 0; ct < 4; ++ct) {
          const v8h blo = *(const v8h*)&Bs[ct * 16 + lr][ks + (grp << 3)];
          const v8h bhi = *(const v8h*)&Bs[ct * 16 + lr][ks + 16 + (grp << 3)];
          const v16h bf = cat8(blo, bhi);
          acc[ct] = __builtin_amdgcn_wmma_f32_16x16x32_f16(false, af, false, bf,
                                                           (short)0, acc[ct],
                                                           false, false);
        }
      }
    }
  }
#pragma unroll
  for (int ct = 0; ct < 4; ++ct) {
    const int col = colBase + ct * 16 + lr;
    const float bv = cpe_b[col];
    _Float16* __restrict__ Dt = D + (size_t)(rowBase + (grp << 3)) * DMODEL + col;
#pragma unroll
    for (int r2 = 0; r2 < 8; ++r2)
      Dt[(size_t)r2 * DMODEL] = (_Float16)(acc[ct][r2] + bv);
  }
}

// ---------------------------------------------------------------------------
// Small GEMM (Ncols=48, f16 operands): one wave per 16x16 tile (x_proj only).
// ---------------------------------------------------------------------------
__global__ void __launch_bounds__(256) gemm_awt_h(
    const _Float16* __restrict__ A, const _Float16* __restrict__ W,
    float* __restrict__ D, int Kdim, int Ncols) {
  const int wave = threadIdx.x >> 5;
  const int lane = threadIdx.x & 31;
  const int grp  = lane >> 4;
  const int lr   = lane & 15;
  const int tiles_n = Ncols >> 4;
  const int tile = blockIdx.x * 8 + wave;
  const int tm = tile / tiles_n;
  const int tn = tile - tm * tiles_n;

  const _Float16* __restrict__ Arow = A + (size_t)((tm << 4) + lr) * Kdim;
  const int col = (tn << 4) + lr;
  const _Float16* __restrict__ Wrow = W + (size_t)col * Kdim;

  v8f acc = {};
  for (int k0 = 0; k0 < Kdim; k0 += 32) {
    __builtin_prefetch(Wrow + k0 + 64, 0, 1);
    const v16h af = cat8(*(const v8h*)(Arow + k0 + (grp << 3)),
                         *(const v8h*)(Arow + k0 + 16 + (grp << 3)));
    const v16h bf = cat8(*(const v8h*)(Wrow + k0 + (grp << 3)),
                         *(const v8h*)(Wrow + k0 + 16 + (grp << 3)));
    acc = __builtin_amdgcn_wmma_f32_16x16x32_f16(false, af, false, bf,
                                                 (short)0, acc, false, false);
  }
  float* __restrict__ Dt = D + (size_t)((tm << 4) + (grp << 3)) * Ncols + col;
#pragma unroll
  for (int r2 = 0; r2 < 8; ++r2) Dt[(size_t)r2 * Ncols] = acc[r2];
}

// ---------------------------------------------------------------------------
// Conversion passes (once per launch).
// ---------------------------------------------------------------------------
__global__ void cvt_f16(const float* __restrict__ src, _Float16* __restrict__ dst,
                        int n) {
  const int g = blockIdx.x * blockDim.x + threadIdx.x;
  if (g < n) dst[g] = (_Float16)src[g];
}
// wt[k][j][c] = w[k][c][j]  (27 x 256 x 256)
__global__ void cvt_tr_cpe(const float* __restrict__ w, _Float16* __restrict__ wt) {
  const int g = blockIdx.x * blockDim.x + threadIdx.x;
  const int k = g >> 16, r = g & 65535, j = r >> 8, c = r & 255;
  wt[g] = (_Float16)w[(k << 16) + (c << 8) + j];
}

// ---------------------------------------------------------------------------
// h = feat + LayerNorm(t2, cnorm_w, cnorm_b). One wave per 256-wide row.
// ---------------------------------------------------------------------------
__global__ void __launch_bounds__(256) cpe_epi(
    const float* __restrict__ t2, const float* __restrict__ feat,
    const float* __restrict__ w, const float* __restrict__ b,
    float* __restrict__ h) {
  const int wave = threadIdx.x >> 5, lane = threadIdx.x & 31;
  const int row = blockIdx.x * 8 + wave;
  const float* __restrict__ x = t2 + (size_t)row * DMODEL;
  float v[8]; float s = 0.f;
#pragma unroll
  for (int j = 0; j < 8; ++j) { v[j] = x[lane + 32 * j]; s += v[j]; }
  const float mu = wave_sum32(s) * (1.0f / DMODEL);
  float q = 0.f;
#pragma unroll
  for (int j = 0; j < 8; ++j) { float d = v[j] - mu; q += d * d; }
  const float rstd = rsqrtf(wave_sum32(q) * (1.0f / DMODEL) + 1e-5f);
#pragma unroll
  for (int j = 0; j < 8; ++j) {
    const int c = lane + 32 * j;
    h[(size_t)row * DMODEL + c] =
        feat[(size_t)row * DMODEL + c] + (v[j] - mu) * rstd * w[c] + b[c];
  }
}

// res = first ? hs : hs + res ; LN(res) -> out32 (residual path) + out16 (GEMM)
__global__ void __launch_bounds__(256) residual_ln(
    const float* __restrict__ hs, float* __restrict__ res,
    const float* __restrict__ w, const float* __restrict__ b,
    float* __restrict__ out32, _Float16* __restrict__ out16, int first) {
  const int wave = threadIdx.x >> 5, lane = threadIdx.x & 31;
  const int row = blockIdx.x * 8 + wave;
  float v[8]; float s = 0.f;
#pragma unroll
  for (int j = 0; j < 8; ++j) {
    const size_t o = (size_t)row * DMODEL + lane + 32 * j;
    float t = hs[o];
    if (!first) t += res[o];
    res[o] = t;
    v[j] = t; s += t;
  }
  const float mu = wave_sum32(s) * (1.0f / DMODEL);
  float q = 0.f;
#pragma unroll
  for (int j = 0; j < 8; ++j) { float d = v[j] - mu; q += d * d; }
  const float rstd = rsqrtf(wave_sum32(q) * (1.0f / DMODEL) + 1e-5f);
#pragma unroll
  for (int j = 0; j < 8; ++j) {
    const int c = lane + 32 * j;
    const float r = (v[j] - mu) * rstd * w[c] + b[c];
    out32[(size_t)row * DMODEL + c] = r;
    out16[(size_t)row * DMODEL + c] = (_Float16)r;
  }
}

// seq[perm[n], c] = src[n, c]
__global__ void scatter_rows(const float* __restrict__ src,
                             const int* __restrict__ perm,
                             float* __restrict__ dst) {
  const int gid = blockIdx.x * blockDim.x + threadIdx.x;
  const int n = gid >> 8, c = gid & 255;
  dst[(size_t)perm[n] * DMODEL + c] = src[gid];
}

// acc[perm[n], c] += src[n, c]   (perm is a permutation -> no conflicts)
__global__ void scatter_add_rows(const float* __restrict__ src,
                                 const int* __restrict__ perm,
                                 float* __restrict__ dst) {
  const int gid = blockIdx.x * blockDim.x + threadIdx.x;
  const int n = gid >> 8, c = gid & 255;
  dst[(size_t)perm[n] * DMODEL + c] += src[gid];
}

// Depthwise causal conv (D_CONV=4) over xz[:, :512] + bias + SiLU.
// Writes f32 (scan input) and f16 (x_proj GEMM input).
__global__ void conv_silu(const float* __restrict__ xz,
                          const float* __restrict__ cw,
                          const float* __restrict__ cb,
                          float* __restrict__ xcs,
                          _Float16* __restrict__ xcsh) {
  const int gid = blockIdx.x * blockDim.x + threadIdx.x;
  const int n = gid >> 9, ch = gid & 511;
  float s = cb[ch];
#pragma unroll
  for (int k = 0; k < 4; ++k) {
    const int m = n - 3 + k;
    if (m >= 0) s += xz[(size_t)m * (2 * DINNER) + ch] * cw[ch * 4 + k];
  }
  const float r = s * sigmoid_(s);
  xcs[gid]  = r;
  xcsh[gid] = (_Float16)r;
}

// dt[n,ch] = softplus( x_dbl[n, :16] . dt_w[ch, :] + dt_b[ch] )
__global__ void dt_softplus(const float* __restrict__ xdbl,
                            const float* __restrict__ dw,
                            const float* __restrict__ db,
                            float* __restrict__ dt) {
  const int gid = blockIdx.x * blockDim.x + threadIdx.x;
  const int n = gid >> 9, ch = gid & 511;
  const float* __restrict__ xr = xdbl + (size_t)n * 48;
  float s = db[ch];
#pragma unroll
  for (int q = 0; q < 16; ++q) s += xr[q] * dw[ch * 16 + q];
  dt[gid] = (s > 20.f) ? s : log1pf(__expf(s));
}

// ---------------------------------------------------------------------------
// Sequential selective scan. One block of 512 threads (16 waves); thread =
// channel, 16 recurrent states in registers; fused z-gate; f16 output
// (feeds out_proj GEMM only).
// ---------------------------------------------------------------------------
__global__ void __launch_bounds__(512) scan_seq(
    const float* __restrict__ dt, const float* __restrict__ xdbl,
    const float* __restrict__ xcs, const float* __restrict__ xz,
    const float* __restrict__ A_log, const float* __restrict__ Dp,
    _Float16* __restrict__ y) {
  const int ch = threadIdx.x;
  float Ar[16];
#pragma unroll
  for (int s = 0; s < 16; ++s) Ar[s] = -__expf(A_log[ch * 16 + s]);
  const float Dv = Dp[ch];
  float st[16];
#pragma unroll
  for (int s = 0; s < 16; ++s) st[s] = 0.f;
  for (int n = 0; n < NPTS; ++n) {
    const float dtv = dt[(size_t)n * DINNER + ch];
    const float xv  = xcs[(size_t)n * DINNER + ch];
    const float zv  = xz[(size_t)n * (2 * DINNER) + DINNER + ch];
    const float* __restrict__ Bm = xdbl + (size_t)n * 48 + 16;
    const float* __restrict__ Cm = xdbl + (size_t)n * 48 + 32;
    const float du = dtv * xv;
    float yv = 0.f;
#pragma unroll
    for (int s = 0; s < 16; ++s) {
      st[s] = __expf(dtv * Ar[s]) * st[s] + du * Bm[s];
      yv += st[s] * Cm[s];
    }
    yv += Dv * xv;
    y[(size_t)n * DINNER + ch] = (_Float16)(yv * (zv * sigmoid_(zv)));
  }
}

__global__ void zero_buf(float* __restrict__ p) {
  p[(size_t)blockIdx.x * blockDim.x + threadIdx.x] = 0.f;
}
__global__ void scale_out(const float* __restrict__ a, float* __restrict__ o) {
  const int gid = blockIdx.x * blockDim.x + threadIdx.x;
  o[gid] = a[gid] * (1.0f / 3.0f);
}

// ---------------------------------------------------------------------------
extern "C" void kernel_launch(void* const* d_in, const int* in_sizes, int n_in,
                              void* d_out, int out_size, void* d_ws, size_t ws_size,
                              hipStream_t stream) {
  const float* feat    = (const float*)d_in[0];
  const float* cpe_w   = (const float*)d_in[1];
  const float* cpe_b   = (const float*)d_in[2];
  const float* fc1_w   = (const float*)d_in[3];
  const float* fc1_b   = (const float*)d_in[4];
  const float* cnorm_w = (const float*)d_in[5];
  const float* cnorm_b = (const float*)d_in[6];
  const float* ln_w    = (const float*)d_in[7];
  const float* ln_b    = (const float*)d_in[8];
  const float* in_w    = (const float*)d_in[9];
  const float* conv_w  = (const float*)d_in[10];
  const float* conv_b  = (const float*)d_in[11];
  const float* xproj_w = (const float*)d_in[12];
  const float* dt_w    = (const float*)d_in[13];
  const float* dt_b    = (const float*)d_in[14];
  const float* A_log   = (const float*)d_in[15];
  const float* Dp      = (const float*)d_in[16];
  const float* out_w   = (const float*)d_in[17];
  const float* normf_w = (const float*)d_in[18];
  const float* normf_b = (const float*)d_in[19];
  const int* nbr_idx   = (const int*)d_in[20];
  const int* nbr_mask  = (const int*)d_in[21];
  const int* orders    = (const int*)d_in[22];
  const int* inverses  = (const int*)d_in[23];
  float* out = (float*)d_out;

  // Workspace layout: f32 region then f16 region.
  float* ws   = (float*)d_ws;
  const size_t Nl = NPTS;
  float* xz   = ws;                 // N*1024
  float* xcs  = xz   + Nl * 1024;   // N*512
  float* dtb  = xcs  + Nl * 512;    // N*512
  float* hbuf = dtb  + Nl * 512;    // N*256  (h after CPE)
  float* fbuf = hbuf + Nl * 256;    // N*256  (seq / hs)
  float* gbuf = fbuf + Nl * 256;    // N*256  (residual)
  float* hln  = gbuf + Nl * 256;    // N*256  (LN out, f32 path)
  float* tmp  = hln  + Nl * 256;    // N*256  (fc1 GEMM out)
  float* acc  = tmp  + Nl * 256;    // N*256  (order accumulator)
  float* xdbl = acc  + Nl * 256;    // N*48
  _Float16* hp    = (_Float16*)(xdbl + Nl * 48);
  _Float16* feath = hp;  hp += Nl * 256;
  _Float16* t1h   = hp;  hp += Nl * 256;
  _Float16* hlnh  = hp;  hp += Nl * 256;
  _Float16* xcsh  = hp;  hp += Nl * 512;
  _Float16* ysh   = hp;  hp += Nl * 512;
  _Float16* cpewt = hp;  hp += (size_t)KNBR * DMODEL * DMODEL;
  _Float16* fc1h  = hp;  hp += DMODEL * DMODEL;
  _Float16* inh   = hp;  hp += (size_t)DEPTH * 1024 * DMODEL;
  _Float16* xpjh  = hp;  hp += (size_t)DEPTH * 48 * DINNER;
  _Float16* outh  = hp;  hp += (size_t)DEPTH * DMODEL * DINNER;

  const dim3 B256(256);

  // ---- one-time f16 mirrors --------------------------------------------
  cvt_f16<<<NPTS * DMODEL / 256, B256, 0, stream>>>(feat, feath, NPTS * DMODEL);
  cvt_tr_cpe<<<KNBR * DMODEL * DMODEL / 256, B256, 0, stream>>>(cpe_w, cpewt);
  cvt_f16<<<DMODEL * DMODEL / 256, B256, 0, stream>>>(fc1_w, fc1h, DMODEL * DMODEL);
  cvt_f16<<<DEPTH * 1024 * DMODEL / 256, B256, 0, stream>>>(in_w, inh,
                                                            DEPTH * 1024 * DMODEL);
  cvt_f16<<<DEPTH * 48 * DINNER / 256, B256, 0, stream>>>(xproj_w, xpjh,
                                                          DEPTH * 48 * DINNER);
  cvt_f16<<<DEPTH * DMODEL * DINNER / 256, B256, 0, stream>>>(out_w, outh,
                                                              DEPTH * DMODEL * DINNER);

  // ---- CPE -------------------------------------------------------------
  cpe_gemm_h<<<dim3(DMODEL / 64, NPTS / 128), B256, 0, stream>>>(
      feath, nbr_idx, nbr_mask, cpewt, cpe_b, t1h);
  gemm_tiled_h<DMODEL><<<dim3(DMODEL / 64, NPTS / 128), B256, 0, stream>>>(
      t1h, fc1h, fc1_b, tmp, DMODEL);
  cpe_epi<<<NPTS / 8, B256, 0, stream>>>(tmp, feat, cnorm_w, cnorm_b, hbuf);
  zero_buf<<<NPTS * DMODEL / 256, B256, 0, stream>>>(acc);

  // ---- Orders ----------------------------------------------------------
  for (int o = 0; o < NORD; ++o) {
    scatter_rows<<<NPTS * DMODEL / 256, B256, 0, stream>>>(
        hbuf, orders + (size_t)o * NPTS, fbuf);
    for (int l = 0; l < DEPTH; ++l) {
      residual_ln<<<NPTS / 8, B256, 0, stream>>>(
          fbuf, gbuf, ln_w + l * DMODEL, ln_b + l * DMODEL, hln, hlnh,
          l == 0 ? 1 : 0);
      // in_proj: (N,256) @ (1024,256)^T -> xz (N,1024)
      gemm_tiled_h<DMODEL><<<dim3(1024 / 64, NPTS / 128), B256, 0, stream>>>(
          hlnh, inh + (size_t)l * 1024 * DMODEL, nullptr, xz, 1024);
      conv_silu<<<NPTS * DINNER / 256, B256, 0, stream>>>(
          xz, conv_w + (size_t)l * DINNER * 4, conv_b + (size_t)l * DINNER,
          xcs, xcsh);
      // x_proj: (N,512) @ (48,512)^T -> xdbl (N,48)
      gemm_awt_h<<<(NPTS / 16) * (48 / 16) / 8, B256, 0, stream>>>(
          xcsh, xpjh + (size_t)l * 48 * DINNER, xdbl, DINNER, 48);
      dt_softplus<<<NPTS * DINNER / 256, B256, 0, stream>>>(
          xdbl, dt_w + (size_t)l * DINNER * 16, dt_b + (size_t)l * DINNER, dtb);
      scan_seq<<<1, 512, 0, stream>>>(
          dtb, xdbl, xcs, xz, A_log + (size_t)l * DINNER * 16,
          Dp + (size_t)l * DINNER, ysh);
      // out_proj: (N,512) @ (256,512)^T -> fbuf (N,256)
      gemm_tiled_h<DINNER><<<dim3(DMODEL / 64, NPTS / 128), B256, 0, stream>>>(
          ysh, outh + (size_t)l * DMODEL * DINNER, nullptr, fbuf, DMODEL);
    }
    residual_ln<<<NPTS / 8, B256, 0, stream>>>(
        fbuf, gbuf, normf_w, normf_b, hln, hlnh, 0);
    scatter_add_rows<<<NPTS * DMODEL / 256, B256, 0, stream>>>(
        hln, inverses + (size_t)o * NPTS, acc);
  }
  scale_out<<<NPTS * DMODEL / 256, B256, 0, stream>>>(acc, out);
  (void)in_sizes; (void)n_in; (void)out_size; (void)ws_size;
}
